// LSTM_42030549958804
// MI455X (gfx1250) — matrix operand order repaired and checked by reading, not compiled, not run on hardware
//
#include <hip/hip_runtime.h>
#include <stddef.h>

// LSTM: B=256, T=512, I=256, H=512, 2 layers + final linear.
// Persistent kernel: 64 WGs x 256 threads = 512 wave32s; wave owns one
// (Mtile, htile) tile = 16 batch rows x 16 hidden cols across all 4 gates.
// Cell state lives in VGPRs. A-panels staged in LDS per block via
// GLOBAL_LOAD_ASYNC_TO_LDS_B128; weights pre-packed in WMMA B-register layout
// and streamed through an explicit 2-stage register pipeline (bounded VGPR
// pressure -> no scratch spills, loads overlap WMMAs).

#define B_   256
#define T_   512
#define I_   256
#define H_   512
#define G4_  2048          // 4*H
#define NKT0 24            // (I+H)/32   layer-0 K tiles
#define NKT1 32            // (H+H)/32   layer-1 K tiles
#define STR0 776           // LDS panel stride layer 0 (768 + 8 pad -> bank stride 4)
#define STR1 1032          // LDS panel stride layer 1 (1024 + 8 pad)
#define NBLK 64
#define NTHR 256

typedef unsigned short bf16s;  // storage type for kernel params

typedef __attribute__((ext_vector_type(16))) __bf16 v16bf;
typedef __attribute__((ext_vector_type(8)))  __bf16 v8bf;
typedef __attribute__((ext_vector_type(8)))  float  v8f;
typedef __attribute__((ext_vector_type(4)))  float  v4f;
typedef __attribute__((ext_vector_type(4)))  int    v4i;

#if defined(__HIP_DEVICE_COMPILE__) && \
    __has_builtin(__builtin_amdgcn_global_load_async_to_lds_b128) && \
    __has_builtin(__builtin_amdgcn_s_wait_asynccnt)
#define USE_ASYNC_LDS 1
#else
#define USE_ASYNC_LDS 0
#endif

__device__ __forceinline__ v8f vzero8() {
  v8f z;
#pragma unroll
  for (int i = 0; i < 8; ++i) z[i] = 0.0f;
  return z;
}

// fast activations via hardware transcendentals (v_exp_f32 / v_rcp_f32)
__device__ __forceinline__ float sigm(float x) {
  return __builtin_amdgcn_rcpf(1.0f + __builtin_amdgcn_exp2f(-1.44269504f * x));
}
__device__ __forceinline__ float tanh_(float x) {
  float e = __builtin_amdgcn_exp2f(2.88539008f * x);  // exp(2x)
  return 1.0f - 2.0f * __builtin_amdgcn_rcpf(e + 1.0f);
}

__device__ __forceinline__ v8f wmma_bf16(v16bf a, v16bf b, v8f c) {
  return __builtin_amdgcn_wmma_f32_16x16x32_bf16(false, a, false, b, (short)0, c,
                                                 false, false);
}

// A-fragment (ISA 16-bit A 16x32 layout): lane<16 holds K=kb+{0..7,16..23},
// lane>=16 holds K=kb+{8..15,24..31}; caller passes k0=kb+laneHi.
__device__ __forceinline__ v16bf a_frag(const __bf16* __restrict__ row, int k0) {
  v8bf lo = *(const v8bf*)(row + k0);
  v8bf hi = *(const v8bf*)(row + k0 + 16);
  v16bf r;
#pragma unroll
  for (int i = 0; i < 8; ++i) { r[i] = lo[i]; r[8 + i] = hi[i]; }
  return r;
}

// load the 4 gate B-tiles for one K-tile; p points at this kt's tile for g=0,
// per-gate displacement is a compile-time immediate.
template <int NKT>
__device__ __forceinline__ void load_b4(const __bf16* p, v16bf b[4]) {
#pragma unroll
  for (int g = 0; g < 4; ++g)
    b[g] = *(const v16bf*)(p + (size_t)g * 32 * NKT * 512);
}

__device__ __forceinline__ void wmma4(v16bf a, const v16bf b[4], v8f acc[4]) {
#pragma unroll
  for (int g = 0; g < 4; ++g) acc[g] = wmma_bf16(a, b[g], acc[g]);
}

// gates[16 x 64] += A[16 x 32*NKT] * B.  A from LDS panel (ds_load_b128),
// B streamed from packed weights through a 2-stage ping-pong pipeline:
// loads for kt+1 are in flight while WMMAs of kt execute; VGPR pressure is
// bounded (2 B-sets + 2 A-frags) so nothing spills.
template <int NKT>
__device__ __forceinline__ void gemm_panel(const __bf16* arow, const __bf16* pWw,
                                           int laneHi, v8f acc[4]) {
  static_assert(NKT >= 4 && (NKT & 1) == 0, "NKT must be even");
  v16bf a0, a1, b0[4], b1[4];
  const __bf16* p = pWw;          // advances 512 elems (1 tile) per kt
  a0 = a_frag(arow, laneHi);
  load_b4<NKT>(p, b0);
  int k0 = laneHi;
#pragma unroll 1
  for (int kt = 0; kt + 2 < NKT; kt += 2) {
    a1 = a_frag(arow, k0 + 32);
    load_b4<NKT>(p + 512, b1);
    wmma4(a0, b0, acc);
    a0 = a_frag(arow, k0 + 64);
    load_b4<NKT>(p + 1024, b0);
    wmma4(a1, b1, acc);
    p += 1024;
    k0 += 64;
  }
  a1 = a_frag(arow, k0 + 32);     // kt = NKT-1
  load_b4<NKT>(p + 512, b1);
  wmma4(a0, b0, acc);             // kt = NKT-2
  wmma4(a1, b1, acc);
}

// stage 16 x 256 fp32 X rows into panel cols [0,256) as bf16 (256 threads)
__device__ __forceinline__ void stage_x_panel(__bf16* panel, const float* __restrict__ X,
                                              int mbase, int t, int tid) {
  const int e = tid * 16;      // 4096 elems total, 16 per thread
  const int row = e >> 8;
  const int col = e & 255;
  const float* src = X + ((size_t)(mbase + row) * T_ + t) * I_ + col;
  __bf16* dst = panel + row * STR0 + col;
#pragma unroll
  for (int i = 0; i < 16; i += 4) {
    v4f v = *(const v4f*)(src + i);
    dst[i + 0] = (__bf16)v[0];
    dst[i + 1] = (__bf16)v[1];
    dst[i + 2] = (__bf16)v[2];
    dst[i + 3] = (__bf16)v[3];
  }
}

// stage 32 x 256 bf16 h rows into panel at column offset colbase
__device__ __forceinline__ void stage_h_panel(__bf16* panel, int stride, int colbase,
                                              const __bf16* __restrict__ hsrc,
                                              int mbase, int tid) {
#pragma unroll
  for (int j = 0; j < 4; ++j) {
    const int c = (tid << 2) + j;   // 1024 x 16B chunks
    const int row = c >> 5;
    const int col = (c & 31) << 3;
    const __bf16* src = hsrc + (size_t)(mbase + row) * H_ + col;
    __bf16* dst = panel + row * stride + colbase + col;
#if USE_ASYNC_LDS
    // GLOBAL_LOAD_ASYNC_TO_LDS_B128 (ASYNCcnt-tracked)
    __builtin_amdgcn_global_load_async_to_lds_b128(
        (__attribute__((address_space(1))) v4i*)src,
        (__attribute__((address_space(3))) v4i*)dst, 0, 0);
#else
    *(v8bf*)dst = *(const v8bf*)src;
#endif
  }
}

__device__ __forceinline__ void panel_commit() {
#if USE_ASYNC_LDS
  __builtin_amdgcn_s_wait_asynccnt(0);
#endif
  __syncthreads();
}

__device__ __forceinline__ void grid_barrier(unsigned* bar, unsigned nb) {
  __syncthreads();
  if (threadIdx.x == 0) {
    volatile unsigned* gen = bar + 64;
    __threadfence();
    unsigned g = *gen;
    unsigned a = atomicAdd(bar, 1u);
    if (a == nb - 1u) {
      *(volatile unsigned*)bar = 0u;
      __threadfence();
      atomicAdd((unsigned*)(bar + 64), 1u);
    } else {
      while (*gen == g) __builtin_amdgcn_s_sleep(2);
    }
  }
  __threadfence();
  __syncthreads();
}

// fused gate activations + cell/hidden update; c stays in registers
__device__ __forceinline__ void lstm_pointwise(const v8f acc[4], const float* __restrict__ bias,
                                               v8f& c, __bf16* __restrict__ hout,
                                               int rbase, int hcol) {
#pragma unroll
  for (int j = 0; j < 8; ++j) {
    float iv = sigm(acc[0][j] + bias[0]);
    float fv = sigm(acc[1][j] + bias[1]);
    float gv = tanh_(acc[2][j] + bias[2]);
    float ov = sigm(acc[3][j] + bias[3]);
    float cc = fv * c[j] + iv * gv;
    c[j] = cc;
    float hv = ov * tanh_(cc);
    hout[(size_t)(rbase + j) * H_ + hcol] = (__bf16)hv;
  }
}

__global__ void __launch_bounds__(NTHR, 1)
lstm_persistent(const float* __restrict__ X,
                const bf16s* __restrict__ pW0s, const bf16s* __restrict__ pW1s,
                const float* __restrict__ bs0, const float* __restrict__ bs1,
                bf16s* __restrict__ h0s, bf16s* __restrict__ h1s,
                const float* __restrict__ Wlin, const float* __restrict__ blin,
                float* __restrict__ y, unsigned* __restrict__ bar) {
  __shared__ __bf16 apanel[16 * STR1];  // 33 KB, reused by both layers

  const __bf16* pW0 = (const __bf16*)pW0s;
  const __bf16* pW1 = (const __bf16*)pW1s;
  __bf16* h0 = (__bf16*)h0s;
  __bf16* h1 = (__bf16*)h1s;

  const int tid   = threadIdx.x;
  const int lane  = tid & 31;
  const int warp  = tid >> 5;
  const int mbase = (blockIdx.x >> 2) * 16;           // shared by all 8 waves
  const int htile = ((blockIdx.x & 3) << 3) + warp;   // 0..31
  const int lrow  = lane & 15;
  const int hcol  = htile * 16 + lrow;
  const int laneHi = (lane < 16) ? 0 : 8;
  const int rbase  = mbase + ((lane < 16) ? 0 : 8);   // C/D row base

  // zero double-buffered h states
  {
    const unsigned stride = gridDim.x * blockDim.x;
    for (unsigned i = blockIdx.x * blockDim.x + tid; i < 2u * B_ * H_; i += stride) {
      h0[i] = (__bf16)0.0f;
      h1[i] = (__bf16)0.0f;
    }
  }

  float b0g[4], b1g[4];
#pragma unroll
  for (int g = 0; g < 4; ++g) {
    b0g[g] = bs0[g * H_ + hcol];
    b1g[g] = bs1[g * H_ + hcol];
  }

  // wave-invariant pointers
  const __bf16* pW0w = pW0 + (size_t)htile * NKT0 * 512 + lane * 16;
  const __bf16* pW1w = pW1 + (size_t)htile * NKT1 * 512 + lane * 16;
  const __bf16* arow0 = apanel + lrow * STR0;
  const __bf16* arow1 = apanel + lrow * STR1;

  v8f c0 = vzero8();  // cell states resident in VGPRs
  v8f c1 = vzero8();

  grid_barrier(bar, gridDim.x);

#pragma unroll 1
  for (int t = 0; t < T_; ++t) {
    const int p = t & 1;
    const __bf16* h0prev = h0 + (size_t)p * B_ * H_;
    __bf16*       h0cur  = h0 + (size_t)(p ^ 1) * B_ * H_;
    const __bf16* h1prev = h1 + (size_t)p * B_ * H_;
    __bf16*       h1cur  = h1 + (size_t)(p ^ 1) * B_ * H_;

    // ---- layer 0: A = [X_t | h0_prev] staged in LDS ----
    stage_x_panel(apanel, X, mbase, t, tid);
    stage_h_panel(apanel, STR0, I_, h0prev, mbase, tid);
    panel_commit();
    {
      v8f acc[4];
#pragma unroll
      for (int g = 0; g < 4; ++g) acc[g] = vzero8();
      gemm_panel<NKT0>(arow0, pW0w, laneHi, acc);
      lstm_pointwise(acc, b0g, c0, h0cur, rbase, hcol);
    }
    __threadfence();
    grid_barrier(bar, gridDim.x);  // entry __syncthreads guards panel reuse

    // ---- layer 1: A = [h0_cur | h1_prev] staged in LDS ----
    stage_h_panel(apanel, STR1, 0, h0cur, mbase, tid);
    stage_h_panel(apanel, STR1, H_, h1prev, mbase, tid);
    panel_commit();
    {
      v8f acc[4];
#pragma unroll
      for (int g = 0; g < 4; ++g) acc[g] = vzero8();
      gemm_panel<NKT1>(arow1, pW1w, laneHi, acc);
      lstm_pointwise(acc, b1g, c1, h1cur, rbase, hcol);
    }
    __threadfence();
    grid_barrier(bar, gridDim.x);
  }

  // final linear: y = h1[T-1] @ W_lin^T + b_lin (last write -> buffer 0)
  if (blockIdx.x == 0) {
    const __bf16* hr = h1 + (size_t)tid * H_;
    float s = blin[0];
    for (int hh = 0; hh < H_; ++hh) s += (float)hr[hh] * Wlin[hh];
    y[tid] = s;
  }
}

// Pre-pack [W_ih | W_hh] (fp32, row-major [4H x K]) into bf16 WMMA B-operand
// layout: tile = 32(K) x 16(N) = 512 elems; (lane, i): n = nt*16 + lane%16,
// k = kt*32 + (lane<16 ? 0 : 8) + (i<8 ? i : i+8). Tiles ordered [nt][kt], kt fastest.
__global__ void pack_weights_kernel(const float* __restrict__ Wih,
                                    const float* __restrict__ Whh, int Kih, int Khh,
                                    bf16s* __restrict__ dsts) {
  __bf16* dst = (__bf16*)dsts;
  const int Ktot = Kih + Khh;
  const int nKt = Ktot >> 5;
  const size_t total = (size_t)G4_ * Ktot;
  size_t idx = (size_t)blockIdx.x * blockDim.x + threadIdx.x;
  if (idx >= total) return;
  int tile = (int)(idx >> 9);
  int r = (int)(idx & 511);
  int lane = r >> 4;
  int i = r & 15;
  int nt = tile / nKt;
  int kt = tile % nKt;
  int n = nt * 16 + (lane & 15);
  int k = kt * 32 + ((lane < 16) ? 0 : 8) + ((i < 8) ? i : i + 8);
  float w = (k < Kih) ? Wih[(size_t)n * Kih + k] : Whh[(size_t)n * Khh + (k - Kih)];
  dst[idx] = (__bf16)w;
}

// fused biases + barrier-state init
__global__ void setup_misc(const float* __restrict__ bih0, const float* __restrict__ bhh0,
                           const float* __restrict__ bih1, const float* __restrict__ bhh1,
                           float* __restrict__ bs0, float* __restrict__ bs1,
                           unsigned* __restrict__ bar) {
  int idx = blockIdx.x * blockDim.x + threadIdx.x;
  if (idx < G4_) bs0[idx] = bih0[idx] + bhh0[idx];
  else if (idx < 2 * G4_) { int n = idx - G4_; bs1[n] = bih1[n] + bhh1[n]; }
  if (idx < 128) bar[idx] = 0u;
}

extern "C" void kernel_launch(void* const* d_in, const int* in_sizes, int n_in,
                              void* d_out, int out_size, void* d_ws, size_t ws_size,
                              hipStream_t stream) {
  (void)in_sizes; (void)n_in; (void)out_size; (void)ws_size;
  const float* X    = (const float*)d_in[0];
  const float* Wih0 = (const float*)d_in[1];
  const float* Whh0 = (const float*)d_in[2];
  const float* bih0 = (const float*)d_in[3];
  const float* bhh0 = (const float*)d_in[4];
  const float* Wih1 = (const float*)d_in[5];
  const float* Whh1 = (const float*)d_in[6];
  const float* bih1 = (const float*)d_in[7];
  const float* bhh1 = (const float*)d_in[8];
  const float* Wlin = (const float*)d_in[9];
  const float* blin = (const float*)d_in[10];
  float* y = (float*)d_out;

  char* ws = (char*)d_ws;
  const size_t off_bar = 0;
  const size_t off_pW0 = 1024;
  const size_t off_pW1 = off_pW0 + (size_t)G4_ * 768 * 2;
  const size_t off_bs0 = off_pW1 + (size_t)G4_ * 1024 * 2;
  const size_t off_bs1 = off_bs0 + (size_t)G4_ * 4;
  const size_t off_h0  = off_bs1 + (size_t)G4_ * 4;
  const size_t off_h1  = off_h0 + 2 * (size_t)B_ * H_ * 2;

  unsigned* bar = (unsigned*)(ws + off_bar);
  bf16s* pW0 = (bf16s*)(ws + off_pW0);
  bf16s* pW1 = (bf16s*)(ws + off_pW1);
  float* bs0 = (float*)(ws + off_bs0);
  float* bs1 = (float*)(ws + off_bs1);
  bf16s* h0  = (bf16s*)(ws + off_h0);
  bf16s* h1  = (bf16s*)(ws + off_h1);

  pack_weights_kernel<<<(G4_ * 768 + NTHR - 1) / NTHR, NTHR, 0, stream>>>(
      Wih0, Whh0, I_, H_, pW0);
  pack_weights_kernel<<<(G4_ * 1024 + NTHR - 1) / NTHR, NTHR, 0, stream>>>(
      Wih1, Whh1, H_, H_, pW1);
  setup_misc<<<16, NTHR, 0, stream>>>(bih0, bhh0, bih1, bhh1, bs0, bs1, bar);

  lstm_persistent<<<NBLK, NTHR, 0, stream>>>(X, pW0, pW1, bs0, bs1, h0, h1, Wlin,
                                             blin, y, bar);
}
